// Varkeys_23158463660655
// MI455X (gfx1250) — matrix operand order repaired and checked by reading, not compiled, not run on hardware
//
#include <hip/hip_runtime.h>

typedef _Float16 half_t;
typedef _Float16 v16h __attribute__((ext_vector_type(16)));
typedef _Float16 v8h  __attribute__((ext_vector_type(8)));
typedef float    v8f  __attribute__((ext_vector_type(8)));

#define B_  4096
#define N_  16384
#define E_  512
#define C_  100
#define CP  128                 // padded C
#define NSPLIT 16
#define NS  (N_ / NSPLIT)       // 1024 keys per block
#define BT  128                 // B rows per block (8 waves * 16)
#define XROW 520                // padded halfword stride of LDS x-tile rows (bank-conflict free)
#define KROW 40                 // padded halfword stride of LDS kern-tile rows

#define XTILE_BYTES (8 * 16 * XROW * 2)   // 133120
#define KTILE_BYTES (8 * 16 * KROW * 2)   // 10240
#define SMEM_BYTES  (XTILE_BYTES + KTILE_BYTES)   // 143360 -> 2 workgroups / WGP

// workspace layout (bytes)
#define WS_XH 0u
#define WS_KH (WS_XH + (size_t)B_ * E_ * 2)        // +4MB
#define WS_VT (WS_KH + (size_t)N_ * E_ * 2)        // +16MB
#define WS_XN (WS_VT + (size_t)CP * N_ * 2)        // +4MB
#define WS_KN (WS_XN + (size_t)B_ * 4)
#define WS_KV (WS_KN + (size_t)N_ * 4)

// ---------------- prep kernels ----------------

__global__ void prep_x(const float* __restrict__ x, half_t* __restrict__ xh,
                       float* __restrict__ xn) {
    int b = blockIdx.x, t = threadIdx.x;   // 128 threads
    float s = 0.f;
    for (int e = t; e < E_; e += 128) {
        float v = x[(size_t)b * E_ + e];
        s += v * v;
        xh[(size_t)b * E_ + e] = (half_t)(-2.0f * v);   // fold the -2 into GEMM1 A
    }
    __shared__ float red[128];
    red[t] = s; __syncthreads();
    for (int off = 64; off > 0; off >>= 1) {
        if (t < off) red[t] += red[t + off];
        __syncthreads();
    }
    if (t == 0) xn[b] = red[0];
}

__global__ void prep_k(const float* __restrict__ k, half_t* __restrict__ kh,
                       float* __restrict__ kn) {
    int n = blockIdx.x, t = threadIdx.x;   // 128 threads
    float s = 0.f;
    for (int e = t; e < E_; e += 128) {
        float v = k[(size_t)n * E_ + e];
        s += v * v;
        kh[(size_t)n * E_ + e] = (half_t)v;
    }
    __shared__ float red[128];
    red[t] = s; __syncthreads();
    for (int off = 64; off > 0; off >>= 1) {
        if (t < off) red[t] += red[t + off];
        __syncthreads();
    }
    if (t == 0) kn[n] = red[0];
}

__global__ void prep_v(const float* __restrict__ val, half_t* __restrict__ vT) {
    int n = blockIdx.x * 256 + threadIdx.x;
    int c = blockIdx.y;                    // 0..127
    vT[(size_t)c * N_ + n] = (c < C_) ? (half_t)val[(size_t)n * C_ + c] : (half_t)0.f;
}

__global__ void zero_kv(float* __restrict__ kv) {
    kv[(size_t)blockIdx.x * 256 + threadIdx.x] = 0.f;
}

// ---------------- fused main kernel ----------------

__global__ __launch_bounds__(256) void varkeys_main(
    const half_t* __restrict__ xh, const half_t* __restrict__ kh,
    const half_t* __restrict__ vT, const float* __restrict__ xn,
    const float* __restrict__ kn, float* __restrict__ kv)
{
    extern __shared__ char smem[];
    const int wave = threadIdx.x >> 5;
    const int lane = threadIdx.x & 31;
    const int lo16 = lane & 15;
    const bool hi  = lane >= 16;

    const int b0      = blockIdx.x * BT + wave * 16;  // this wave's 16 B rows
    const int n_begin = blockIdx.y * NS;

    half_t* xt = (half_t*)smem + (size_t)wave * 16 * XROW;
    half_t* kt = (half_t*)(smem + XTILE_BYTES) + (size_t)wave * 16 * KROW;

    // Stage this wave's 16 x-rows (f16, prescaled by -2) into LDS.
    for (int i = lane; i < 16 * 64; i += 32) {
        int r = i >> 6, cblk = (i & 63) * 8;
        *(v8h*)(xt + (size_t)r * XROW + cblk) =
            *(const v8h*)(xh + (size_t)(b0 + r) * E_ + cblk);
    }
    __syncthreads();

    // Per-lane ||x||^2 + 1 for the 8 D-matrix rows this lane holds.
    const int mbase = hi ? 8 : 0;
    float xnr[8];
#pragma unroll
    for (int r = 0; r < 8; ++r) xnr[r] = xn[b0 + mbase + r] + 1.0f;

    v8f acc[8] = {};                       // KV accumulators: 8 C-tiles of 16 cols

    const int aoff = hi ? 16 : 0;          // A-operand lane K offset
    const int boff = hi ? 16 : 0;          // B-operand lane K offset

    for (int nc = 0; nc < NS; nc += 32) {
        const int n0 = n_begin + nc;
        if (nc + 32 < NS) {                // prefetch next chunk
            __builtin_prefetch(kh + (size_t)(n0 + 32 + lo16) * E_, 0, 1);
            __builtin_prefetch(vT + (size_t)(lane * 4) * N_ + n0 + 32, 0, 1);
        }

        // ---- GEMM1: d[16x32] = (-2x)[16xE] . keys^T ; C preloaded with xn+kn+1
        const float kn0 = kn[n0 + lo16];
        const float kn1 = kn[n0 + 16 + lo16];
        v8f d0, d1;
#pragma unroll
        for (int r = 0; r < 8; ++r) { d0[r] = xnr[r] + kn0; d1[r] = xnr[r] + kn1; }

        const half_t* k0 = kh + (size_t)(n0 + lo16) * E_ + boff;
        const half_t* k1 = kh + (size_t)(n0 + 16 + lo16) * E_ + boff;
#pragma unroll 4
        for (int e = 0; e < E_; e += 32) {
            v8h a0 = *(const v8h*)(xt + (size_t)lo16 * XROW + e + aoff);
            v8h a1 = *(const v8h*)(xt + (size_t)lo16 * XROW + e + aoff + 8);
            v16h A = __builtin_shufflevector(a0, a1, 0,1,2,3,4,5,6,7,8,9,10,11,12,13,14,15);
            v16h Bm0 = *(const v16h*)(k0 + e);
            d0 = __builtin_amdgcn_wmma_f32_16x16x32_f16(false, A, false, Bm0, (short)0, d0, false, false);
            v16h Bm1 = *(const v16h*)(k1 + e);
            d1 = __builtin_amdgcn_wmma_f32_16x16x32_f16(false, A, false, Bm1, (short)0, d1, false, false);
        }

        // kern = 1/(d+1) via v_rcp_f32; transpose D-layout -> A-layout through LDS.
#pragma unroll
        for (int r = 0; r < 8; ++r) {
            kt[(size_t)(mbase + r) * KROW +      lo16] = (half_t)__builtin_amdgcn_rcpf(d0[r]);
            kt[(size_t)(mbase + r) * KROW + 16 + lo16] = (half_t)__builtin_amdgcn_rcpf(d1[r]);
        }

        v8h c0 = *(const v8h*)(kt + (size_t)lo16 * KROW + aoff);
        v8h c1 = *(const v8h*)(kt + (size_t)lo16 * KROW + aoff + 8);
        v16h KA = __builtin_shufflevector(c0, c1, 0,1,2,3,4,5,6,7,8,9,10,11,12,13,14,15);

        // ---- GEMM2: acc[ct] += kern[16x32] . valuesT^T[32x16]
#pragma unroll
        for (int ct = 0; ct < 8; ++ct) {
            v16h Bv = *(const v16h*)(vT + (size_t)(ct * 16 + lo16) * N_ + n0 + boff);
            acc[ct] = __builtin_amdgcn_wmma_f32_16x16x32_f16(false, KA, false, Bv, (short)0, acc[ct], false, false);
        }
    }

    // Accumulate this N-slice's partial KV into the f32 workspace.
#pragma unroll
    for (int ct = 0; ct < 8; ++ct) {
#pragma unroll
        for (int r = 0; r < 8; ++r) {
            float* p = kv + (size_t)(b0 + mbase + r) * CP + ct * 16 + lo16;
            __hip_atomic_fetch_add(p, acc[ct][r], __ATOMIC_RELAXED, __HIP_MEMORY_SCOPE_AGENT);
        }
    }
}

// ---------------- row-normalize ----------------

__global__ void normalize_kernel(const float* __restrict__ kv, float* __restrict__ out) {
    int b = blockIdx.x, t = threadIdx.x;   // 128 threads
    float v = (t < C_) ? kv[(size_t)b * CP + t] : 0.f;
    __shared__ float red[128];
    red[t] = v; __syncthreads();
    for (int off = 64; off > 0; off >>= 1) {
        if (t < off) red[t] += red[t + off];
        __syncthreads();
    }
    float s = red[0];
    if (t < C_) out[(size_t)b * C_ + t] = v / s;
}

// ---------------- launcher ----------------

extern "C" void kernel_launch(void* const* d_in, const int* in_sizes, int n_in,
                              void* d_out, int out_size, void* d_ws, size_t ws_size,
                              hipStream_t stream) {
    const float* x      = (const float*)d_in[0];
    const float* keys   = (const float*)d_in[1];
    const float* values = (const float*)d_in[2];
    float* out = (float*)d_out;

    char* ws = (char*)d_ws;
    half_t* xh = (half_t*)(ws + WS_XH);
    half_t* kh = (half_t*)(ws + WS_KH);
    half_t* vT = (half_t*)(ws + WS_VT);
    float*  xn = (float*)(ws + WS_XN);
    float*  kn = (float*)(ws + WS_KN);
    float*  kv = (float*)(ws + WS_KV);

    (void)hipFuncSetAttribute((const void*)varkeys_main,
                              hipFuncAttributeMaxDynamicSharedMemorySize, SMEM_BYTES);

    prep_x<<<B_, 128, 0, stream>>>(x, xh, xn);
    prep_k<<<N_, 128, 0, stream>>>(keys, kh, kn);
    prep_v<<<dim3(N_ / 256, CP), 256, 0, stream>>>(values, vT);
    zero_kv<<<(B_ * CP) / 256, 256, 0, stream>>>(kv);
    varkeys_main<<<dim3(B_ / BT, NSPLIT), 256, SMEM_BYTES, stream>>>(xh, kh, vT, xn, kn, kv);
    normalize_kernel<<<B_, 128, 0, stream>>>(kv, out);
}
